// GCNContextAggregator_20040317403439
// MI455X (gfx1250) — matrix operand order repaired and checked by reading, not compiled, not run on hardware
//
#include <hip/hip_runtime.h>
#include <cstdint>
#include <cstddef>

#define IN_F   1536
#define HID    384
#define NNODES 50000
#define NEDGES 1600000
#define CATF   (IN_F + 3 * HID)   // 2688

typedef __attribute__((ext_vector_type(16))) __bf16         v16bf;
typedef __attribute__((ext_vector_type(8)))  float          v8f;
typedef __attribute__((ext_vector_type(16))) unsigned short ushort16;
typedef __attribute__((ext_vector_type(8)))  unsigned short ushort8;

// ---------- helpers ----------
__device__ __forceinline__ unsigned short f2bf(float f) {
  unsigned int u = __builtin_bit_cast(unsigned int, f);
  u += 0x7FFFu + ((u >> 16) & 1u);           // round-to-nearest-even
  return (unsigned short)(u >> 16);
}

__device__ __forceinline__ float gelu_exact(float x) {
  return 0.5f * x * (1.0f + erff(x * 0.70710678118654752f));
}

// CDNA5 async global->LDS copy (16B per lane), tracked by ASYNCcnt
__device__ __forceinline__ void async_copy_b128(unsigned lds_off, const void* gptr) {
  asm volatile("global_load_async_to_lds_b128 %0, %1, off"
               :: "v"(lds_off), "v"((unsigned long long)(uintptr_t)gptr)
               : "memory");
}
__device__ __forceinline__ void wait_async0() {
  asm volatile("s_wait_asynccnt 0x0" ::: "memory");
}

// ---------- WMMA GEMM: C[M,HID] (+)= A_bf16[M,K] @ W[K,HID] ----------
// A is pre-converted bf16 row-major [M,K]. Bt is W transposed to [HID, ldb] bf16.
// B panel (64 cols x 128 K) staged in LDS via global_load_async_to_lds_b128 and
// shared by all 8 waves of the block. Wave computes a 16x64 C strip:
// one A fragment feeds 4 WMMAs per K-step. grid=(ceil(M/128), HID/64), block=256.
__global__ __launch_bounds__(256)
void wmma_gemm_abf16(const unsigned short* __restrict__ A,
                     const unsigned short* __restrict__ Bt,
                     float* __restrict__ C, int M, int K, int ldb, int accflag)
{
  __shared__ unsigned short Bs[64 * 128];     // 16 KB bf16: [col][k]

  const int tid  = threadIdx.x;
  const int lane = tid & 31;
  const int wave = tid >> 5;
  const int lm   = lane & 15;
  const int hi   = lane >> 4;                 // 0: lanes 0-15, 1: lanes 16-31
  const int rowTile = blockIdx.x * 128 + wave * 16;
  const int colBase = blockIdx.y * 64;

  int arow = rowTile + lm;                    // clamp: EXEC stays all-ones
  if (arow >= M) arow = M - 1;
  const unsigned short* Ar = A + (size_t)arow * K;

  // cooperative B loader: 64 rows x 256B; thread -> (row rbase+16j, 16B unit sub)
  const int sub   = tid & 15;
  const int rbase = tid >> 4;

  v8f c0 = {}, c1 = {}, c2 = {}, c3 = {};

  for (int k0 = 0; k0 < K; k0 += 128) {
    __syncthreads();                          // previous chunk fully consumed
#pragma unroll
    for (int j = 0; j < 4; ++j) {
      int rr = rbase + j * 16;                // 0..63
      const unsigned short* src = Bt + (size_t)(colBase + rr) * ldb + k0 + sub * 8;
      async_copy_b128((unsigned)(uintptr_t)(Bs + rr * 128 + sub * 8), src);
    }
    wait_async0();                            // this wave's async copies done
    __syncthreads();                          // all waves' copies visible

#pragma unroll
    for (int kk = 0; kk < 128; kk += 32) {
      // A 16-bit fragment: lo lanes K+0..15, hi lanes K+16..31 (contiguous, matches B)
      const unsigned short* ap = Ar + k0 + kk + hi * 16;
      ushort8 alo = *(const ushort8*)(ap);
      ushort8 ahi = *(const ushort8*)(ap + 8);
      v16bf a = __builtin_bit_cast(v16bf,
                  __builtin_shufflevector(alo, ahi,
                    0, 1, 2, 3, 4, 5, 6, 7, 8, 9, 10, 11, 12, 13, 14, 15));
      // B fragments from LDS: col = colBase + t*16 + lm, contiguous 16 bf16
      const unsigned short* bp = Bs + kk + hi * 16;
      v16bf b0 = __builtin_bit_cast(v16bf, *(const ushort16*)(bp + (0 * 16 + lm) * 128));
      v16bf b1 = __builtin_bit_cast(v16bf, *(const ushort16*)(bp + (1 * 16 + lm) * 128));
      v16bf b2 = __builtin_bit_cast(v16bf, *(const ushort16*)(bp + (2 * 16 + lm) * 128));
      v16bf b3 = __builtin_bit_cast(v16bf, *(const ushort16*)(bp + (3 * 16 + lm) * 128));
      c0 = __builtin_amdgcn_wmma_f32_16x16x32_bf16(false, a, false, b0, (short)0, c0, false, false);
      c1 = __builtin_amdgcn_wmma_f32_16x16x32_bf16(false, a, false, b1, (short)0, c1, false, false);
      c2 = __builtin_amdgcn_wmma_f32_16x16x32_bf16(false, a, false, b2, (short)0, c2, false, false);
      c3 = __builtin_amdgcn_wmma_f32_16x16x32_bf16(false, a, false, b3, (short)0, c3, false, false);
    }
  }

  // C/D layout: VGPR r -> row rowTile + r + 8*hi, col colBase + t*16 + (lane&15)
  const int srow = rowTile + hi * 8;
#pragma unroll
  for (int r = 0; r < 8; ++r) {
    int rr = srow + r;
    if (rr < M) {
      float* crow = C + (size_t)rr * HID + colBase + lm;
      if (accflag) {
        crow[0] += c0[r]; crow[16] += c1[r]; crow[32] += c2[r]; crow[48] += c3[r];
      } else {
        crow[0]  = c0[r]; crow[16]  = c1[r]; crow[32]  = c2[r]; crow[48]  = c3[r];
      }
    }
  }
}

// ---------- weight transpose + bf16 convert: W[K,N] f32 -> Wt[N,K] bf16 ----------
__global__ void convert_wt(const float* __restrict__ W, unsigned short* __restrict__ Wt,
                           int K, int N)
{
  size_t i = (size_t)blockIdx.x * blockDim.x + threadIdx.x;
  if (i >= (size_t)K * N) return;
  int k = (int)(i / N), n = (int)(i % N);
  Wt[(size_t)n * K + k] = f2bf(W[i]);
}

// ---------- elementwise f32 -> bf16 ----------
__global__ void convert_bf16(const float* __restrict__ src, unsigned short* __restrict__ dst,
                             size_t total)
{
  size_t i = (size_t)blockIdx.x * blockDim.x + threadIdx.x;
  if (i < total) dst[i] = f2bf(src[i]);
}

// ---------- degree / norm ----------
__global__ void deg_init(float* deg, int n) {
  int i = blockIdx.x * blockDim.x + threadIdx.x;
  if (i < n) deg[i] = 1.0f;                    // self loop
}
__global__ void deg_edges(const long long* __restrict__ ei, float* deg, int nE) {
  int e = blockIdx.x * blockDim.x + threadIdx.x;
  if (e < nE) atomicAdd(&deg[(int)ei[nE + e]], 1.0f);
}
__global__ void deg_rsqrt(float* deg, int n) {
  int i = blockIdx.x * blockDim.x + threadIdx.x;
  if (i < n) deg[i] = rsqrtf(deg[i]);
}

// ---------- self-loop init: agg = hw * dinv^2 ----------
__global__ void self_init(const float* __restrict__ hw, const float* __restrict__ dinv,
                          float* __restrict__ agg, int n)
{
  size_t i = (size_t)blockIdx.x * blockDim.x + threadIdx.x;
  if (i >= (size_t)n * HID) return;
  float di = dinv[i / HID];
  agg[i] = hw[i] * di * di;
}

// ---------- edge scatter: agg[dst] += hw[src] * dinv[src]*dinv[dst] ----------
// wave32 per edge, 12 f32 atomics per lane; hw & agg (76.8MB each) are L2-resident
__global__ __launch_bounds__(256)
void scatter_edges(const long long* __restrict__ ei, const float* __restrict__ hw,
                   const float* __restrict__ dinv, float* __restrict__ agg, int nE)
{
  int e = blockIdx.x * 8 + (int)(threadIdx.x >> 5);
  if (e >= nE) return;
  int lane = threadIdx.x & 31;
  int s = (int)ei[e];
  int d = (int)ei[nE + e];
  float nrm = dinv[s] * dinv[d];
  const float* hs = hw + (size_t)s * HID;
  float* ad = agg + (size_t)d * HID;
#pragma unroll
  for (int j = 0; j < HID / 32; ++j) {
    int c = lane + j * 32;
    atomicAdd(&ad[c], hs[c] * nrm);
  }
}

// ---------- bias + GELU, fp32 agg -> bf16 activation ----------
__global__ void bias_gelu_bf16(const float* __restrict__ agg, const float* __restrict__ b,
                               unsigned short* __restrict__ h, int n)
{
  size_t i = (size_t)blockIdx.x * blockDim.x + threadIdx.x;
  if (i >= (size_t)n * HID) return;
  h[i] = f2bf(gelu_exact(agg[i] + b[i % HID]));
}

// ---------- final: out = LayerNorm(GELU(z + mlp_b)) * g + beta, wave per row ----------
__global__ __launch_bounds__(256)
void gelu_ln(const float* __restrict__ z, const float* __restrict__ mb,
             const float* __restrict__ g, const float* __restrict__ be,
             float* __restrict__ out, int n)
{
  int row = blockIdx.x * 8 + (int)(threadIdx.x >> 5);
  if (row >= n) return;
  int lane = threadIdx.x & 31;
  const float* zr = z + (size_t)row * HID;
  float v[HID / 32];
  float sum = 0.0f;
#pragma unroll
  for (int j = 0; j < HID / 32; ++j) {
    int c = lane + j * 32;
    float x = gelu_exact(zr[c] + mb[c]);
    v[j] = x; sum += x;
  }
#pragma unroll
  for (int o = 16; o > 0; o >>= 1) sum += __shfl_xor(sum, o, 32);
  float mu = sum * (1.0f / HID);
  float vs = 0.0f;
#pragma unroll
  for (int j = 0; j < HID / 32; ++j) { float d = v[j] - mu; vs += d * d; }
#pragma unroll
  for (int o = 16; o > 0; o >>= 1) vs += __shfl_xor(vs, o, 32);
  float inv = rsqrtf(vs * (1.0f / HID) + 1e-5f);
  float* orow = out + (size_t)row * HID;
#pragma unroll
  for (int j = 0; j < HID / 32; ++j) {
    int c = lane + j * 32;
    orow[c] = (v[j] - mu) * inv * g[c] + be[c];
  }
}

// ---------- host side ----------
extern "C" void kernel_launch(void* const* d_in, const int* in_sizes, int n_in,
                              void* d_out, int out_size, void* d_ws, size_t ws_size,
                              hipStream_t stream)
{
  (void)in_sizes; (void)n_in; (void)out_size; (void)ws_size;
  const float*      x   = (const float*)d_in[0];
  const long long*  ei  = (const long long*)d_in[1];
  const float*      W0  = (const float*)d_in[2];
  const float*      b0  = (const float*)d_in[3];
  const float*      W1  = (const float*)d_in[4];
  const float*      b1  = (const float*)d_in[5];
  const float*      W2  = (const float*)d_in[6];
  const float*      b2  = (const float*)d_in[7];
  const float*      mW  = (const float*)d_in[8];
  const float*      mb  = (const float*)d_in[9];
  const float*      g   = (const float*)d_in[10];
  const float*      be  = (const float*)d_in[11];
  float*            out = (float*)d_out;

  // workspace partition
  char*  ws  = (char*)d_ws;
  size_t off = 0;
  auto alloc = [&](size_t bytes) -> char* {
    off = (off + 255) & ~(size_t)255;
    char* p = ws + off; off += bytes; return p;
  };
  float*          dinv = (float*)alloc((size_t)NNODES * 4);
  unsigned short* Wt0  = (unsigned short*)alloc((size_t)HID * IN_F * 2);
  unsigned short* Wt1  = (unsigned short*)alloc((size_t)HID * HID * 2);
  unsigned short* Wt2  = (unsigned short*)alloc((size_t)HID * HID * 2);
  unsigned short* mWt  = (unsigned short*)alloc((size_t)HID * CATF * 2);
  unsigned short* xbf  = (unsigned short*)alloc((size_t)NNODES * IN_F * 2);  // 153.6MB
  unsigned short* hP   = (unsigned short*)alloc((size_t)NNODES * HID * 2);   // 38.4MB
  unsigned short* hQ   = (unsigned short*)alloc((size_t)NNODES * HID * 2);   // 38.4MB
  float*          hw   = (float*)alloc((size_t)NNODES * HID * 4);            // 76.8MB
  float*          agg  = (float*)alloc((size_t)NNODES * HID * 4);            // 76.8MB
  float*          z    = (float*)alloc((size_t)NNODES * HID * 4);            // 76.8MB

  const size_t nh = (size_t)NNODES * HID;
  const size_t nx = (size_t)NNODES * IN_F;
  const int    TB = 256;
  dim3 gemmGrid((NNODES + 127) / 128, HID / 64);
  int  ewBlocks = (NEDGES + 7) / 8;              // wave-per-edge
  int  nhBlocks = (int)((nh + TB - 1) / TB);
  int  nxBlocks = (int)((nx + TB - 1) / TB);
  int  lnBlocks = (NNODES + 7) / 8;

  // 1) prep: weights (transposed bf16) and x in bf16
  convert_wt<<<(IN_F * HID + TB - 1) / TB, TB, 0, stream>>>(W0, Wt0, IN_F, HID);
  convert_wt<<<(HID * HID + TB - 1) / TB, TB, 0, stream>>>(W1, Wt1, HID, HID);
  convert_wt<<<(HID * HID + TB - 1) / TB, TB, 0, stream>>>(W2, Wt2, HID, HID);
  convert_wt<<<(CATF * HID + TB - 1) / TB, TB, 0, stream>>>(mW, mWt, CATF, HID);
  convert_bf16<<<nxBlocks, TB, 0, stream>>>(x, xbf, nx);

  // 2) symmetric GCN normalization (deg includes self loop)
  deg_init<<<(NNODES + TB - 1) / TB, TB, 0, stream>>>(dinv, NNODES);
  deg_edges<<<(NEDGES + TB - 1) / TB, TB, 0, stream>>>(ei, dinv, NEDGES);
  deg_rsqrt<<<(NNODES + TB - 1) / TB, TB, 0, stream>>>(dinv, NNODES);

  // 3) z = x @ mlp_W[0:1536]  (concat chunk fused; cat never materialized)
  wmma_gemm_abf16<<<gemmGrid, TB, 0, stream>>>(xbf, mWt + 0, z, NNODES, IN_F, CATF, 0);

  // 4) hop 1: hw = x@W0 ; aggregate ; gelu(+b0) -> hP (bf16)
  wmma_gemm_abf16<<<gemmGrid, TB, 0, stream>>>(xbf, Wt0, hw, NNODES, IN_F, IN_F, 0);
  self_init<<<nhBlocks, TB, 0, stream>>>(hw, dinv, agg, NNODES);
  scatter_edges<<<ewBlocks, TB, 0, stream>>>(ei, hw, dinv, agg, NEDGES);
  bias_gelu_bf16<<<nhBlocks, TB, 0, stream>>>(agg, b0, hP, NNODES);
  wmma_gemm_abf16<<<gemmGrid, TB, 0, stream>>>(hP, mWt + IN_F, z, NNODES, HID, CATF, 1);

  // 5) hop 2: hw = h1@W1 ; aggregate ; gelu(+b1) -> hQ
  wmma_gemm_abf16<<<gemmGrid, TB, 0, stream>>>(hP, Wt1, hw, NNODES, HID, HID, 0);
  self_init<<<nhBlocks, TB, 0, stream>>>(hw, dinv, agg, NNODES);
  scatter_edges<<<ewBlocks, TB, 0, stream>>>(ei, hw, dinv, agg, NEDGES);
  bias_gelu_bf16<<<nhBlocks, TB, 0, stream>>>(agg, b1, hQ, NNODES);
  wmma_gemm_abf16<<<gemmGrid, TB, 0, stream>>>(hQ, mWt + IN_F + HID, z, NNODES, HID, CATF, 1);

  // 6) hop 3: hw = h2@W2 ; aggregate ; gelu(+b2) -> hP (h1 dead)
  wmma_gemm_abf16<<<gemmGrid, TB, 0, stream>>>(hQ, Wt2, hw, NNODES, HID, HID, 0);
  self_init<<<nhBlocks, TB, 0, stream>>>(hw, dinv, agg, NNODES);
  scatter_edges<<<ewBlocks, TB, 0, stream>>>(ei, hw, dinv, agg, NEDGES);
  bias_gelu_bf16<<<nhBlocks, TB, 0, stream>>>(agg, b2, hP, NNODES);
  wmma_gemm_abf16<<<gemmGrid, TB, 0, stream>>>(hP, mWt + IN_F + 2 * HID, z, NNODES, HID, CATF, 1);

  // 7) out = LN(GELU(z + mlp_b))
  gelu_ln<<<lnBlocks, TB, 0, stream>>>(z, mb, g, be, out, NNODES);
}
